// MultiHeadAttention_52871047414119
// MI455X (gfx1250) — matrix operand order repaired and checked by reading, not compiled, not run on hardware
//
#include <hip/hip_runtime.h>
#include <hip/hip_bf16.h>

// ---------------------------------------------------------------------------
// Multi-head causal attention forward on gfx1250 (CDNA5, wave32, WMMA bf16).
// B=4, T=2048, D=512, H=8, DH=64.
//   - all GEMM-like math on v_wmma_f32_16x16x32_bf16 (f32 accumulate)
//   - K/V tiles staged with GLOBAL_LOAD_ASYNC_TO_LDS_B128 + double buffering
//     (falls back to synchronous staging if the builtin is unavailable)
// ---------------------------------------------------------------------------

typedef __attribute__((ext_vector_type(16))) __bf16 v16bf;
typedef __attribute__((ext_vector_type(8)))  float  v8f;
typedef int v4i __attribute__((vector_size(16)));   // matches builtin param type

#define Bc 4
#define Tc 2048
#define Dc 512
#define Hc 8
#define DHc 64

#if defined(__gfx1250__) && __has_builtin(__builtin_amdgcn_global_load_async_to_lds_b128)
#define USE_ASYNC_LDS 1
#endif

static __device__ __forceinline__ void async_ld128(const void* gsrc, void* ldst) {
#if defined(USE_ASYNC_LDS)
  __builtin_amdgcn_global_load_async_to_lds_b128(
      (__attribute__((address_space(1))) v4i*)gsrc,
      (__attribute__((address_space(3))) v4i*)ldst, 0, 0);
#else
  *(uint4*)ldst = *(const uint4*)gsrc;
#endif
}

static __device__ __forceinline__ void async_wait0() {
#if defined(USE_ASYNC_LDS)
#if __has_builtin(__builtin_amdgcn_s_wait_asynccnt)
  __builtin_amdgcn_s_wait_asynccnt(0);
#else
  asm volatile("s_wait_asynccnt 0x0" ::: "memory");
#endif
#endif
}

static __device__ __forceinline__ unsigned short f2bf(float f) {
  unsigned u = __float_as_uint(f);
  u += 0x7FFFu + ((u >> 16) & 1u);            // round-to-nearest-even
  return (unsigned short)(u >> 16);
}

// A-fragment (16x32 bf16, ISA 7.12.2): lane L holds row M=L&15.
// lanes 0-15: K = 0..7 and 16..23 ; lanes 16-31: K = 8..15 and 24..31.
static __device__ __forceinline__ v16bf ld_frag_a(const unsigned short* base,
                                                  int row0, int rowstride,
                                                  int koff, int lane) {
  int m  = row0 + (lane & 15);
  int kb = (lane >> 4) * 8;
  const unsigned short* p = base + m * rowstride + koff + kb;
  union { v16bf v; uint4 u[2]; } r;
  r.u[0] = *(const uint4*)(p);
  r.u[1] = *(const uint4*)(p + 16);
  return r.v;
}

// B-fragment (32x16 bf16): lane L holds col N=L&15; lanes 0-15 K=0..7 and 16..23,
// lanes 16-31 K=8..15 and 24..31 (LDS tile is N-major).
static __device__ __forceinline__ v16bf ld_frag_b(const unsigned short* base,
                                                  int col0, int rowstride,
                                                  int koff, int lane) {
  int n  = col0 + (lane & 15);
  int kb = (lane >> 4) * 8;
  const unsigned short* p = base + n * rowstride + koff + kb;
  union { v16bf v; uint4 u[2]; } r;
  r.u[0] = *(const uint4*)(p);
  r.u[1] = *(const uint4*)(p + 16);
  return r.v;
}

static __device__ __forceinline__ v8f wmma_bf16(v16bf a, v16bf b, v8f c) {
  return __builtin_amdgcn_wmma_f32_16x16x32_bf16(false, a, false, b,
                                                 (short)0, c, false, false);
}

// ---------------------------------------------------------------------------
__global__ __launch_bounds__(256) void cvt_f32_bf16(const float* __restrict__ in,
                                                    unsigned short* __restrict__ out,
                                                    int n) {
  int i = blockIdx.x * 256 + threadIdx.x;
  if (i < n) out[i] = f2bf(in[i]);
}

// ---------------------------------------------------------------------------
// Tiled bf16 GEMM: C[M,N] = A[M,K] * W[K,N].  Block tile 128x128, 8 waves in
// a 4x2 grid, wave tile 32x64 (2x4 WMMA accumulators), K staged 32 at a time.
// A tile staged via async-to-LDS (overlaps with B's load+transpose).
// MODE 0: QKV epilogue -> scatter Q (x 1/8), K, V(transposed) as bf16.
// MODE 1: plain fp32 store to fout (out-projection).
template <int MODE>
__global__ __launch_bounds__(256) void gemm_bf16_wmma(
    const unsigned short* __restrict__ A, const unsigned short* __restrict__ W,
    int lda, int ldb, int Ktot,
    unsigned short* __restrict__ qout, unsigned short* __restrict__ kout,
    unsigned short* __restrict__ vtout, float* __restrict__ fout, int ldc) {
  __shared__ __align__(16) unsigned short ldsA[128 * 32];   // row-major, K contig
  __shared__ __align__(16) unsigned short ldsB[128 * 32];   // N-major, K contig

  const int tid  = threadIdx.x;
  const int lane = tid & 31;
  const int w    = tid >> 5;
  const int m0   = blockIdx.x * 128;
  const int n0g  = blockIdx.y * 128;
  const int m_off = (w >> 1) * 32;
  const int n_off = (w & 1) * 64;

  v8f acc[2][4] = {};

  for (int kc = 0; kc < Ktot; kc += 32) {
    // --- stage A tile 128x32 async (512 x 16B chunks, 2 per thread) ---
#pragma unroll
    for (int it = 0; it < 2; ++it) {
      int c = tid + it * 256;
      int r = c >> 2, p = c & 3;
      async_ld128(A + (size_t)(m0 + r) * lda + kc + p * 8, ldsA + r * 32 + p * 8);
    }
    // --- stage B tile 32x128, transposed into N-major LDS ---
    {
      int kk  = tid >> 3;
      int n16 = (tid & 7) * 16;
      union { uint4 u; unsigned short s[8]; } t0, t1;
      t0.u = *(const uint4*)(W + (size_t)(kc + kk) * ldb + n0g + n16);
      t1.u = *(const uint4*)(W + (size_t)(kc + kk) * ldb + n0g + n16 + 8);
#pragma unroll
      for (int e = 0; e < 8; ++e) {
        ldsB[(n16 + e) * 32 + kk]     = t0.s[e];
        ldsB[(n16 + 8 + e) * 32 + kk] = t1.s[e];
      }
    }
    if (kc + 32 < Ktot)
      __builtin_prefetch(W + (size_t)(kc + 32) * ldb + n0g, 0, 1);
    async_wait0();
    __syncthreads();

    v16bf a0 = ld_frag_a(ldsA, m_off,      32, 0, lane);
    v16bf a1 = ld_frag_a(ldsA, m_off + 16, 32, 0, lane);
    v16bf bfr[4];
#pragma unroll
    for (int j = 0; j < 4; ++j)
      bfr[j] = ld_frag_b(ldsB, n_off + j * 16, 32, 0, lane);
#pragma unroll
    for (int j = 0; j < 4; ++j) {
      acc[0][j] = wmma_bf16(a0, bfr[j], acc[0][j]);
      acc[1][j] = wmma_bf16(a1, bfr[j], acc[1][j]);
    }
    __syncthreads();
  }

  // --- epilogue: C layout = VGPR e -> row (+e | +8+e by lane half), N = lane&15
  const int rbase = (lane >> 4) * 8;
#pragma unroll
  for (int i = 0; i < 2; ++i)
#pragma unroll
    for (int j = 0; j < 4; ++j)
#pragma unroll
      for (int e = 0; e < 8; ++e) {
        int grow = m0 + m_off + i * 16 + rbase + e;          // global M (b*T+t)
        int c    = n0g + n_off + j * 16 + (lane & 15);       // global N
        float v  = acc[i][j][e];
        if (MODE == 0) {
          int b = grow >> 11, t = grow & (Tc - 1);
          if (c < Dc) {                                      // Q, scaled 1/sqrt(DH)
            int h = c >> 6, dh = c & 63;
            qout[((((size_t)b * Hc + h) * Tc + t) << 6) + dh] = f2bf(v * 0.125f);
          } else if (c < 2 * Dc) {                           // K: [B,H,T,DH]
            int cc = c - Dc, h = cc >> 6, dh = cc & 63;
            kout[((((size_t)b * Hc + h) * Tc + t) << 6) + dh] = f2bf(v);
          } else {                                           // V transposed: [B,H,DH,T]
            int cc = c - 2 * Dc, h = cc >> 6, dh = cc & 63;
            vtout[((((size_t)b * Hc + h) << 6) + dh) * Tc + t] = f2bf(v);
          }
        } else {
          fout[(size_t)grow * ldc + c] = v;
        }
      }
}

// ---------------------------------------------------------------------------
// Flash attention: grid (T/128, B*H); 8 waves, each owns 16 query rows.
// K/V tiles double-buffered: next tile's async copy issued before computing
// on the current tile (copy/compute overlap via ASYNCcnt).
__global__ __launch_bounds__(256) void attn_wmma(
    const unsigned short* __restrict__ q, const unsigned short* __restrict__ k,
    const unsigned short* __restrict__ vt, unsigned short* __restrict__ attnb) {
  __shared__ __align__(16) unsigned short ldsK[2][64 * 64];   // [key][dh]
  __shared__ __align__(16) unsigned short ldsV[2][64 * 64];   // [dh][key]
  __shared__ __align__(16) unsigned short ldsP[8 * 16 * 64];

  const int tid  = threadIdx.x;
  const int lane = tid & 31;
  const int w    = tid >> 5;
  const int qtile = blockIdx.x;
  const int bh    = blockIdx.y;
  const int b = bh >> 3, h = bh & 7;

  const unsigned short* qp  = q  + (size_t)bh * Tc * DHc;
  const unsigned short* kp  = k  + (size_t)bh * Tc * DHc;
  const unsigned short* vtp = vt + (size_t)bh * DHc * Tc;

  const int q0 = qtile * 128 + w * 16;

  // Q A-fragments, loaded once straight from global (row-major [t][dh]).
  v16bf aq0 = ld_frag_a(qp, q0, DHc, 0,  lane);
  v16bf aq1 = ld_frag_a(qp, q0, DHc, 32, lane);

  v8f acc_o[4] = {};
  float m8[8], l8[8];
#pragma unroll
  for (int e = 0; e < 8; ++e) { m8[e] = -1e30f; l8[e] = 0.0f; }

  const int rbase = (lane >> 4) * 8;
  const float LOG2E = 1.4426950408889634f;
  const int kb_end = qtile * 2 + 2;                 // causal: keys <= q0+127

  // stage K (64x64, row=key) and V (64x64, row=dh) tiles into buffer `buf`
  auto stage = [&](int kb, int buf) {
#pragma unroll
    for (int it = 0; it < 2; ++it) {
      int c = tid + it * 256;
      int r = c >> 3, p = c & 7;
      async_ld128(kp + (size_t)(kb * 64 + r) * DHc + p * 8,
                  &ldsK[buf][r * 64 + p * 8]);
      async_ld128(vtp + (size_t)r * Tc + kb * 64 + p * 8,
                  &ldsV[buf][r * 64 + p * 8]);
    }
  };

  stage(0, 0);
  async_wait0();
  __syncthreads();

  for (int kb = 0; kb < kb_end; ++kb) {
    const int cur = kb & 1;
    if (kb + 1 < kb_end) stage(kb + 1, cur ^ 1);    // overlap copy with compute

    if (kb * 64 <= q0 + 15) {                       // some key unmasked for wave
      const unsigned short* Kb = ldsK[cur];
      const unsigned short* Vb = ldsV[cur];
      // preload all K B-fragments, then issue 8 WMMAs back-to-back
      v16bf bk0[4], bk1[4];
#pragma unroll
      for (int j = 0; j < 4; ++j) {
        bk0[j] = ld_frag_b(Kb, j * 16, DHc, 0,  lane);
        bk1[j] = ld_frag_b(Kb, j * 16, DHc, 32, lane);
      }
      v8f s[4];
#pragma unroll
      for (int j = 0; j < 4; ++j) s[j] = wmma_bf16(aq0, bk0[j], (v8f){});
#pragma unroll
      for (int j = 0; j < 4; ++j) s[j] = wmma_bf16(aq1, bk1[j], s[j]);

      // causal mask + online softmax (rows live across 16-lane halves)
      int col0 = kb * 64 + (lane & 15);
#pragma unroll
      for (int e = 0; e < 8; ++e) {
        int row = q0 + rbase + e;
        float mx = -1e30f;
#pragma unroll
        for (int j = 0; j < 4; ++j) {
          if (col0 + j * 16 > row) s[j][e] = -1e30f;
          mx = fmaxf(mx, s[j][e]);
        }
#pragma unroll
        for (int d = 1; d < 16; d <<= 1) mx = fmaxf(mx, __shfl_xor(mx, d, 32));
        float mn = fmaxf(m8[e], mx);
        float alpha = exp2f((m8[e] - mn) * LOG2E);
        float rs = 0.0f;
#pragma unroll
        for (int j = 0; j < 4; ++j) {
          float pv = exp2f((s[j][e] - mn) * LOG2E);
          s[j][e] = pv;
          rs += pv;
        }
#pragma unroll
        for (int d = 1; d < 16; d <<= 1) rs += __shfl_xor(rs, d, 32);
        l8[e] = l8[e] * alpha + rs;
#pragma unroll
        for (int jd = 0; jd < 4; ++jd) acc_o[jd][e] *= alpha;
        m8[e] = mn;
      }
      // P -> per-wave LDS (bf16, row-major [16][64]); same-wave DS is in-order
      unsigned short* pw = ldsP + w * 16 * 64;
#pragma unroll
      for (int j = 0; j < 4; ++j)
#pragma unroll
        for (int e = 0; e < 8; ++e)
          pw[(rbase + e) * 64 + j * 16 + (lane & 15)] = f2bf(s[j][e]);

      // O += P * V : preload fragments, then 8 WMMAs back-to-back
      v16bf ap0 = ld_frag_a(pw, 0, 64, 0,  lane);
      v16bf ap1 = ld_frag_a(pw, 0, 64, 32, lane);
      v16bf bv0[4], bv1[4];
#pragma unroll
      for (int jd = 0; jd < 4; ++jd) {
        bv0[jd] = ld_frag_b(Vb, jd * 16, 64, 0,  lane);
        bv1[jd] = ld_frag_b(Vb, jd * 16, 64, 32, lane);
      }
#pragma unroll
      for (int jd = 0; jd < 4; ++jd) acc_o[jd] = wmma_bf16(ap0, bv0[jd], acc_o[jd]);
#pragma unroll
      for (int jd = 0; jd < 4; ++jd) acc_o[jd] = wmma_bf16(ap1, bv1[jd], acc_o[jd]);
    }
    async_wait0();
    __syncthreads();
  }

  // epilogue: O/l -> attn buffer in [B,T,D] bf16 (h-major concat)
#pragma unroll
  for (int e = 0; e < 8; ++e) {
    float inv = 1.0f / l8[e];
    int t = q0 + rbase + e;
#pragma unroll
    for (int jd = 0; jd < 4; ++jd) {
      int dh = jd * 16 + (lane & 15);
      attnb[((size_t)(b * Tc + t) << 9) + h * DHc + dh] = f2bf(acc_o[jd][e] * inv);
    }
  }
}

// ---------------------------------------------------------------------------
extern "C" void kernel_launch(void* const* d_in, const int* in_sizes, int n_in,
                              void* d_out, int out_size, void* d_ws, size_t ws_size,
                              hipStream_t stream) {
  (void)in_sizes; (void)n_in; (void)out_size; (void)ws_size;
  const float* x    = (const float*)d_in[0];   // [B,T,D]
  const float* wqkv = (const float*)d_in[1];   // [D,3D]
  const float* wout = (const float*)d_in[2];   // [D,D]
  float* out = (float*)d_out;                  // [B,T,D]

  char* ws = (char*)d_ws;
  const size_t N_X  = (size_t)Bc * Tc * Dc;         // 4,194,304
  const size_t N_WQ = (size_t)Dc * 3 * Dc;          //   786,432
  const size_t N_WO = (size_t)Dc * Dc;              //   262,144
  const size_t N_HD = (size_t)Bc * Hc * Tc * DHc;   // 4,194,304

  unsigned short* xb    = (unsigned short*)(ws);
  unsigned short* wqkvb = (unsigned short*)(ws + 2 * N_X);
  unsigned short* woutb = (unsigned short*)(ws + 2 * (N_X + N_WQ));
  unsigned short* qb    = (unsigned short*)(ws + 2 * (N_X + N_WQ + N_WO));
  unsigned short* kb    = qb + N_HD;
  unsigned short* vtb   = kb + N_HD;
  unsigned short* attnb = vtb + N_HD;

  cvt_f32_bf16<<<(int)((N_X  + 255) / 256), 256, 0, stream>>>(x,    xb,    (int)N_X);
  cvt_f32_bf16<<<(int)((N_WQ + 255) / 256), 256, 0, stream>>>(wqkv, wqkvb, (int)N_WQ);
  cvt_f32_bf16<<<(int)((N_WO + 255) / 256), 256, 0, stream>>>(wout, woutb, (int)N_WO);

  // QKV projection: [8192,512] x [512,1536]
  gemm_bf16_wmma<0><<<dim3(64, 12), 256, 0, stream>>>(
      xb, wqkvb, Dc, 3 * Dc, Dc, qb, kb, vtb, nullptr, 0);

  // flash attention: (T/128, B*H)
  attn_wmma<<<dim3(Tc / 128, Bc * Hc), 256, 0, stream>>>(qb, kb, vtb, attnb);

  // out projection: [8192,512] x [512,512] -> fp32
  gemm_bf16_wmma<1><<<dim3(64, 4), 256, 0, stream>>>(
      attnb, woutb, Dc, Dc, Dc, nullptr, nullptr, nullptr, out, Dc);
}
